// SkipEncoder_62758062129855
// MI455X (gfx1250) — matrix operand order repaired and checked by reading, not compiled, not run on hardware
//
#include <hip/hip_runtime.h>
#include <hip/hip_bf16.h>
#include <stdint.h>

typedef _Float16 f16;
typedef __attribute__((ext_vector_type(16))) _Float16 v16h;
typedef __attribute__((ext_vector_type(8)))  float    v8f;
typedef __attribute__((ext_vector_type(4)))  unsigned int v4u;
typedef __attribute__((ext_vector_type(8)))  int      v8i;
typedef __attribute__((ext_vector_type(4)))  int      v4i;

union Frag16 { v16h v; uint32_t u[8]; };

__device__ __forceinline__ v8f wmma_f16(v16h a, v16h b, v8f c) {
  // D = A(16x32 f16) * B(32x16 f16) + C(16x16 f32)
  return __builtin_amdgcn_wmma_f32_16x16x32_f16(false, a, false, b, (short)0, c,
                                                false, false);
}

// A fragment: 16x32 f16 tile from row-major [M][K] storage (ld in halves).
// ISA 7.12.2: lanes0-15 hold M=lane, K={0..7,16..23}; lanes16-31 K={8..15,24..31}
__device__ __forceinline__ v16h load_A(const f16* base, int row0, int k0, int ld,
                                       int lane) {
  const int m = lane & 15;
  const int kh = (lane < 16) ? 0 : 8;
  const f16* p = base + (row0 + m) * ld + k0 + kh;
  Frag16 f;
#pragma unroll
  for (int i = 0; i < 4; ++i) f.u[i] = *(const uint32_t*)(p + 2 * i);
#pragma unroll
  for (int i = 0; i < 4; ++i) f.u[4 + i] = *(const uint32_t*)(p + 16 + 2 * i);
  return f.v;
}

// B fragment: 32x16 (KxN) from row-major [N][K] storage (weights are [out][in]).
// lanes0-15: N=lane, K={0..7,16..23} ; lanes16-31: N=lane-16, K={8..15,24..31}
__device__ __forceinline__ v16h load_B(const f16* base, int n0, int k0, int ld,
                                       int lane) {
  const int n = lane & 15;
  const int ko = (lane < 16) ? 0 : 8;
  const f16* p = base + (n0 + n) * ld + k0 + ko;
  Frag16 f;
#pragma unroll
  for (int i = 0; i < 4; ++i) f.u[i] = *(const uint32_t*)(p + 2 * i);
#pragma unroll
  for (int i = 0; i < 4; ++i) f.u[4 + i] = *(const uint32_t*)(p + 16 + 2 * i);
  return f.v;
}

// ---- Tensor Data Mover: 3D tile (x=8, y=8, z=64ch) of f32 ----------------
// D# per CDNA5 ISA.  strides: row=256, channel=65536.
__device__ __forceinline__ void tdm_groups(uint64_t ga, uint32_t lds_off,
                                           v4u* g0, v8i* g1, v4i* g2, v4i* g3) {
  v4u a;
  a.x = 1u;                                            // count=1 (valid user D#)
  a.y = lds_off;                                       // lds_addr (bytes)
  a.z = (uint32_t)ga;                                  // global_addr[31:0]
  a.w = (uint32_t)((ga >> 32) & 0x01FFFFFFu) | (2u << 30);  // [56:32] | type=2
  v8i b;
  b[0] = 0x00020000;          // workgroup_mask=0, data_size=2 (4 bytes)
  b[1] = (int)(8u << 16);     // tensor_dim0 = 8 (bits 79:48, low half here)
  b[2] = (int)(8u << 16);     // tensor_dim0 hi | tensor_dim1 = 8
  b[3] = (int)(8u << 16);     // tensor_dim1 hi | tile_dim0 = 8
  b[4] = (int)(8u | (64u << 16)); // tile_dim1 = 8 | tile_dim2 = 64
  b[5] = 256;                 // tensor_dim0_stride = 256 elements (row of x)
  b[6] = 0;                   // stride0 hi | stride1 lo16 (65536 -> 0)
  b[7] = 1;                   // tensor_dim1_stride >> 16  (65536)
  v4i c; c[0] = 64; c[1] = 0; c[2] = 0; c[3] = 0;      // tensor_dim2 = 64
  v4i d = {0, 0, 0, 0};
  *g0 = a; *g1 = b; *g2 = c; *g3 = d;
}

__device__ __forceinline__ void tdm_load_tile(const void* gp, uint32_t lds_off) {
  v4u g0; v8i g1; v4i g2, g3;
  tdm_groups((uint64_t)(uintptr_t)gp, lds_off, &g0, &g1, &g2, &g3);
  v8i g4 = {0, 0, 0, 0, 0, 0, 0, 0};
  __builtin_amdgcn_tensor_load_to_lds(g0, g1, g2, g3, g4, 0);
}

__device__ __forceinline__ void tdm_store_tile(void* gp, uint32_t lds_off) {
  v4u g0; v8i g1; v4i g2, g3;
  tdm_groups((uint64_t)(uintptr_t)gp, lds_off, &g0, &g1, &g2, &g3);
  v8i g4 = {0, 0, 0, 0, 0, 0, 0, 0};
  __builtin_amdgcn_tensor_store_from_lds(g0, g1, g2, g3, g4, 0);
}

// ---- geometry ----
#define EMB 128
#define CIN 64
#define T 64          // tokens per 8x8 window
#define NWIN 8192     // 8 * 32 * 32

// LDS leading dims (halves / floats), all even so 32-bit pair loads stay aligned
#define LDX 72        // x tile  [64 tok][64 ch] f16
#define LDW 132       // xw      [64][128] f32 (residual stream)
#define LDH 136       // f16 activations [64][128]
#define LDA 68        // attn    [h][64][64] f16
#define LDV 72        // vT      [128][64] f16

#define OFF_XW   0                       // 64*132*4 = 33792  f32 residual
#define OFF_XW16 33792                   // 64*136*2 = 17408  f16 act (A input)
#define OFF_QO   (33792+17408)           // 17408: TDM x-tile f32(16K), sQ, sO
#define OFF_KOUT (OFF_QO+17408)          // 17408: sK, later packed out tile f32
#define OFF_VT   (OFF_KOUT+17408)        // 128*72*2 = 18432
#define OFF_XA   (OFF_VT+18432)          // max(xtile f16 9216, attn 34816)
#define SMEM_BYTES (OFF_XA+34816)        // = 139264

__global__ void __launch_bounds__(256)
cvt_w_f16(const float* __restrict__ src, f16* __restrict__ dst, int n) {
  int i = blockIdx.x * 256 + threadIdx.x;
  if (i < n) dst[i] = (f16)src[i];
}

__global__ void __launch_bounds__(256)
swin_block_fused(const float* __restrict__ x,
                 const f16*   __restrict__ wPin,  const float* __restrict__ bPin,
                 const f16*   __restrict__ wQKV,  const float* __restrict__ bQKV,
                 const f16*   __restrict__ wAO,   const float* __restrict__ bAO,
                 const f16*   __restrict__ wMLP,  const float* __restrict__ bMLP,
                 const float* __restrict__ g1, const float* __restrict__ b1,
                 const float* __restrict__ g2, const float* __restrict__ b2,
                 const f16*   __restrict__ wPO,   const float* __restrict__ bPO,
                 float* __restrict__ out) {
  __shared__ __align__(256) char smem[SMEM_BYTES];
  float* sXwF  = (float*)(smem + OFF_XW);
  f16*   sXw16 = (f16*)(smem + OFF_XW16);
  const float* sXf32 = (const float*)(smem + OFF_QO);  // TDM-landed [c][t] f32
  f16*   sQ    = (f16*)(smem + OFF_QO);
  f16*   sO    = (f16*)(smem + OFF_QO);
  f16*   sK    = (f16*)(smem + OFF_KOUT);
  float* sOutT = (float*)(smem + OFF_KOUT);            // packed [oc][t] f32
  f16*   sVT   = (f16*)(smem + OFF_VT);
  f16*   sX16  = (f16*)(smem + OFF_XA);
  f16*   sAttn = (f16*)(smem + OFF_XA);

  const int tid  = threadIdx.x;
  const int lane = tid & 31;
  const int wv   = tid >> 5;           // 8 waves
  const int nfr  = lane & 15;          // C/B fragment column within tile
  const int mb   = (lane < 16) ? 0 : 8;

  const int w  = blockIdx.x;
  const int b  = w >> 10;              // /1024
  const int wh = (w >> 5) & 31;
  const int ww = w & 31;

  const float* xs = x + (size_t)b * (64 * 65536) + (size_t)(wh * 8) * 256 + ww * 8;
  float* os = out + (size_t)b * (64 * 65536) + (size_t)(wh * 8) * 256 + ww * 8;

  // ------- Stage A0: TDM DMA of the x window tile into LDS ([c][t] f32) ---
  if (wv == 0) {
    tdm_load_tile(xs, (uint32_t)(uintptr_t)(smem + OFF_QO));
    __builtin_amdgcn_s_wait_tensorcnt(0);
  }
  __syncthreads();

  // ------- Stage A1: convert to f16 [t][c] for WMMA A fragments -----------
  for (int j = tid; j < 1024; j += 256) {
    int c = j >> 4;                 // channel
    int t0 = (j & 15) * 4;          // token
    float4 v = *((const float4*)sXf32 + j);
    f16* d = sX16 + t0 * LDX + c;
    d[0 * LDX] = (f16)v.x; d[1 * LDX] = (f16)v.y;
    d[2 * LDX] = (f16)v.z; d[3 * LDX] = (f16)v.w;
  }
  __syncthreads();

  // ------------- proj_in: xw[64,128] = xtile[64,64] @ Wpin^T + b ----------
  {
    const int mt = wv >> 1, ntb = (wv & 1) * 4;
    v16h af[2];
#pragma unroll
    for (int kt = 0; kt < 2; ++kt) af[kt] = load_A(sX16, mt * 16, kt * 32, LDX, lane);
#pragma unroll
    for (int nt = 0; nt < 4; ++nt) {
      const int n0 = (ntb + nt) * 16;
      v8f acc = {};
#pragma unroll
      for (int kt = 0; kt < 2; ++kt)
        acc = wmma_f16(af[kt], load_B(wPin, n0, kt * 32, CIN, lane), acc);
      const float bias = bPin[n0 + nfr];
#pragma unroll
      for (int r = 0; r < 8; ++r) {
        float v = acc[r] + bias;
        int t = mt * 16 + mb + r, e = n0 + nfr;
        sXwF[t * LDW + e]  = v;
        sXw16[t * LDH + e] = (f16)v;
      }
    }
  }
  __syncthreads();

  // ------------- qkv: [64,384] = xw16 @ Wqkv^T + b ; route q/k/vT --------
  {
    const int ntb = wv * 3;
#pragma unroll
    for (int j = 0; j < 3; ++j) {
      const int n0 = (ntb + j) * 16;
      v16h bf[4];                            // cache W fragments across mt
#pragma unroll
      for (int kt = 0; kt < 4; ++kt) bf[kt] = load_B(wQKV, n0, kt * 32, EMB, lane);
#pragma unroll
      for (int mt = 0; mt < 4; ++mt) {
        v8f acc = {};
#pragma unroll
        for (int kt = 0; kt < 4; ++kt)
          acc = wmma_f16(load_A(sXw16, mt * 16, kt * 32, LDH, lane), bf[kt], acc);
        const float bias = bQKV[n0 + nfr];
#pragma unroll
        for (int r = 0; r < 8; ++r) {
          float v = acc[r] + bias;
          int t = mt * 16 + mb + r;
          if (n0 < 128)       sQ[t * LDH + n0 + nfr]           = (f16)v;
          else if (n0 < 256)  sK[t * LDH + (n0 - 128) + nfr]   = (f16)v;
          else                sVT[((n0 - 256) + nfr) * LDV + t] = (f16)v;
        }
      }
    }
  }
  __syncthreads();

  // ------- scores (q@k^T)*scale, softmax, store attn f16 (wave-local) -----
  const int h   = wv >> 1;          // head
  const int mt0 = (wv & 1) * 2;     // this wave owns rows [mt0*16, mt0*16+32)
  {
    const float scale = 0.17677669529663687f;
    v8f s[2][4];
#pragma unroll
    for (int m2 = 0; m2 < 2; ++m2) {
      v16h a = load_A(sQ, (mt0 + m2) * 16, h * 32, LDH, lane);
#pragma unroll
      for (int nt = 0; nt < 4; ++nt) {
        v8f z = {};
        s[m2][nt] = wmma_f16(a, load_B(sK, nt * 16, h * 32, LDH, lane), z);
      }
    }
#pragma unroll
    for (int m2 = 0; m2 < 2; ++m2) {
#pragma unroll
      for (int r = 0; r < 8; ++r) {
        float mx = -3.0e38f;
#pragma unroll
        for (int nt = 0; nt < 4; ++nt) {
          s[m2][nt][r] *= scale;
          mx = fmaxf(mx, s[m2][nt][r]);
        }
        mx = fmaxf(mx, __shfl_xor(mx, 1));   // masks 1,2,4,8 keep each 16-lane
        mx = fmaxf(mx, __shfl_xor(mx, 2));   // half separate -> per-row reduce
        mx = fmaxf(mx, __shfl_xor(mx, 4));
        mx = fmaxf(mx, __shfl_xor(mx, 8));
        float sum = 0.f;
#pragma unroll
        for (int nt = 0; nt < 4; ++nt) {
          float p = __expf(s[m2][nt][r] - mx);
          s[m2][nt][r] = p;
          sum += p;
        }
        sum += __shfl_xor(sum, 1);
        sum += __shfl_xor(sum, 2);
        sum += __shfl_xor(sum, 4);
        sum += __shfl_xor(sum, 8);
        const float inv = 1.0f / sum;
        const int t = (mt0 + m2) * 16 + mb + r;
#pragma unroll
        for (int nt = 0; nt < 4; ++nt)
          sAttn[h * (T * LDA) + t * LDA + nt * 16 + nfr] = (f16)(s[m2][nt][r] * inv);
      }
    }
  }
  // no barrier: attn & this wave's sQ/sO block are wave-local; vT sync'd above

  // --------------- o = attn @ v  (per head, wave-local rows) --------------
  {
    v16h bf[2][2];
#pragma unroll
    for (int nt = 0; nt < 2; ++nt)
#pragma unroll
      for (int kt = 0; kt < 2; ++kt)
        bf[nt][kt] = load_B(sVT, h * 32 + nt * 16, kt * 32, LDV, lane);
#pragma unroll
    for (int m2 = 0; m2 < 2; ++m2) {
      v16h af[2];
#pragma unroll
      for (int kt = 0; kt < 2; ++kt)
        af[kt] = load_A(sAttn + h * (T * LDA), (mt0 + m2) * 16, kt * 32, LDA, lane);
#pragma unroll
      for (int nt = 0; nt < 2; ++nt) {
        v8f acc = {};
#pragma unroll
        for (int kt = 0; kt < 2; ++kt) acc = wmma_f16(af[kt], bf[nt][kt], acc);
#pragma unroll
        for (int r = 0; r < 8; ++r) {
          int t = (mt0 + m2) * 16 + mb + r;
          sO[t * LDH + h * 32 + nt * 16 + nfr] = (f16)acc[r];
        }
      }
    }
  }
  __syncthreads();

  // -------- attn_out proj + residual into sXwF ---------------------------
  {
    const int mt = wv >> 1, ntb = (wv & 1) * 4;
    v16h af[4];
#pragma unroll
    for (int kt = 0; kt < 4; ++kt) af[kt] = load_A(sO, mt * 16, kt * 32, LDH, lane);
#pragma unroll
    for (int nt = 0; nt < 4; ++nt) {
      const int n0 = (ntb + nt) * 16;
      v8f acc = {};
#pragma unroll
      for (int kt = 0; kt < 4; ++kt)
        acc = wmma_f16(af[kt], load_B(wAO, n0, kt * 32, EMB, lane), acc);
      const float bias = bAO[n0 + nfr];
#pragma unroll
      for (int r = 0; r < 8; ++r) {
        int t = mt * 16 + mb + r, e = n0 + nfr;
        sXwF[t * LDW + e] += acc[r] + bias;
      }
    }
  }
  __syncthreads();

  // --------------------------- LayerNorm 1 -------------------------------
  {
    const int row = tid >> 2, part = tid & 3;
    float* pr = sXwF + row * LDW + part * 32;
    float s1 = 0.f, s2 = 0.f;
#pragma unroll
    for (int i = 0; i < 32; ++i) { float v = pr[i]; s1 += v; s2 += v * v; }
    s1 += __shfl_xor(s1, 1); s1 += __shfl_xor(s1, 2);
    s2 += __shfl_xor(s2, 1); s2 += __shfl_xor(s2, 2);
    const float mu = s1 * (1.0f / 128.0f);
    const float rs = rsqrtf(s2 * (1.0f / 128.0f) - mu * mu + 1e-5f);
#pragma unroll
    for (int i = 0; i < 32; ++i) {
      int e = part * 32 + i;
      float y = (pr[i] - mu) * rs * g1[e] + b1[e];
      pr[i] = y;
      sXw16[row * LDH + e] = (f16)y;
    }
  }
  __syncthreads();

  // ------------------ MLP + residual into sXwF ---------------------------
  {
    const int mt = wv >> 1, ntb = (wv & 1) * 4;
    v16h af[4];
#pragma unroll
    for (int kt = 0; kt < 4; ++kt) af[kt] = load_A(sXw16, mt * 16, kt * 32, LDH, lane);
#pragma unroll
    for (int nt = 0; nt < 4; ++nt) {
      const int n0 = (ntb + nt) * 16;
      v8f acc = {};
#pragma unroll
      for (int kt = 0; kt < 4; ++kt)
        acc = wmma_f16(af[kt], load_B(wMLP, n0, kt * 32, EMB, lane), acc);
      const float bias = bMLP[n0 + nfr];
#pragma unroll
      for (int r = 0; r < 8; ++r) {
        int t = mt * 16 + mb + r, e = n0 + nfr;
        sXwF[t * LDW + e] += acc[r] + bias;
      }
    }
  }
  __syncthreads();

  // --------------------------- LayerNorm 2 -------------------------------
  {
    const int row = tid >> 2, part = tid & 3;
    float* pr = sXwF + row * LDW + part * 32;
    float s1 = 0.f, s2 = 0.f;
#pragma unroll
    for (int i = 0; i < 32; ++i) { float v = pr[i]; s1 += v; s2 += v * v; }
    s1 += __shfl_xor(s1, 1); s1 += __shfl_xor(s1, 2);
    s2 += __shfl_xor(s2, 1); s2 += __shfl_xor(s2, 2);
    const float mu = s1 * (1.0f / 128.0f);
    const float rs = rsqrtf(s2 * (1.0f / 128.0f) - mu * mu + 1e-5f);
#pragma unroll
    for (int i = 0; i < 32; ++i) {
      int e = part * 32 + i;
      sXw16[row * LDH + e] = (f16)((pr[i] - mu) * rs * g2[e] + b2[e]);
    }
  }
  __syncthreads();

  // ----- proj_out: [64,64] = y @ Wpo^T + b ; packed [oc][t] for TDM ------
  {
    const int mt = wv >> 1, ntb = (wv & 1) * 2;
    v16h af[4];
#pragma unroll
    for (int kt = 0; kt < 4; ++kt) af[kt] = load_A(sXw16, mt * 16, kt * 32, LDH, lane);
#pragma unroll
    for (int nt = 0; nt < 2; ++nt) {
      const int n0 = (ntb + nt) * 16;
      v8f acc = {};
#pragma unroll
      for (int kt = 0; kt < 4; ++kt)
        acc = wmma_f16(af[kt], load_B(wPO, n0, kt * 32, EMB, lane), acc);
      const float bias = bPO[n0 + nfr];
#pragma unroll
      for (int r = 0; r < 8; ++r) {
        int t = mt * 16 + mb + r;
        sOutT[(n0 + nfr) * 64 + t] = acc[r] + bias;  // packed [oc][t]
      }
    }
  }
  __syncthreads();

  // ------- TDM DMA store: packed [oc][r][col] tile -> out[b,oc,h,w] -------
  if (wv == 0) {
    tdm_store_tile(os, (uint32_t)(uintptr_t)(smem + OFF_KOUT));
    __builtin_amdgcn_s_wait_tensorcnt(0);  // (s_endpgm also waits-idle)
  }
}

// d_ws layout (halves): [0) inProj 49152 | attnOut 16384 | mlp 16384
//                       | projOut 8192 | projIn 8192 ]  = 196608 bytes
extern "C" void kernel_launch(void* const* d_in, const int* in_sizes, int n_in,
                              void* d_out, int out_size, void* d_ws, size_t ws_size,
                              hipStream_t stream) {
  const float* x          = (const float*)d_in[0];
  const float* proj_in_w  = (const float*)d_in[1];
  const float* proj_in_b  = (const float*)d_in[2];
  const float* in_proj_w  = (const float*)d_in[3];
  const float* in_proj_b  = (const float*)d_in[4];
  const float* attn_out_w = (const float*)d_in[5];
  const float* attn_out_b = (const float*)d_in[6];
  const float* mlp_w      = (const float*)d_in[7];
  const float* mlp_b      = (const float*)d_in[8];
  const float* ln1_g      = (const float*)d_in[9];
  const float* ln1_b      = (const float*)d_in[10];
  const float* ln2_g      = (const float*)d_in[11];
  const float* ln2_b      = (const float*)d_in[12];
  const float* proj_out_w = (const float*)d_in[13];
  const float* proj_out_b = (const float*)d_in[14];

  f16* wsH = (f16*)d_ws;
  f16* wQKV = wsH;              // 384*128
  f16* wAO  = wsH + 49152;      // 128*128
  f16* wMLP = wsH + 65536;      // 128*128
  f16* wPO  = wsH + 81920;      // 64*128
  f16* wPin = wsH + 90112;      // 128*64

  cvt_w_f16<<<(49152 + 255) / 256, 256, 0, stream>>>(in_proj_w,  wQKV, 49152);
  cvt_w_f16<<<(16384 + 255) / 256, 256, 0, stream>>>(attn_out_w, wAO,  16384);
  cvt_w_f16<<<(16384 + 255) / 256, 256, 0, stream>>>(mlp_w,      wMLP, 16384);
  cvt_w_f16<<<(8192  + 255) / 256, 256, 0, stream>>>(proj_out_w, wPO,  8192);
  cvt_w_f16<<<(8192  + 255) / 256, 256, 0, stream>>>(proj_in_w,  wPin, 8192);

  swin_block_fused<<<NWIN, 256, 0, stream>>>(
      x, wPin, proj_in_b, wQKV, in_proj_b, wAO, attn_out_b, wMLP, mlp_b,
      ln1_g, ln1_b, ln2_g, ln2_b, wPO, proj_out_b, (float*)d_out);
}
